// DeepRNNNetwork_43894565765265
// MI455X (gfx1250) — matrix operand order repaired and checked by reading, not compiled, not run on hardware
//
#include <hip/hip_runtime.h>
#include <hip/hip_bf16.h>

// CDNA5 wave32 WMMA types
typedef __bf16 v16bf __attribute__((ext_vector_type(16)));
typedef float  v8f   __attribute__((ext_vector_type(8)));

#define WMMA_BF16(a, b, c) \
  __builtin_amdgcn_wmma_f32_16x16x32_bf16(false, (a), false, (b), (short)0, (c), false, false)

// ---------------------------------------------------------------------------
// B-matrix fragment (32x16 KxN, bf16) from a row-major f32 weight matrix W[n][k]
// for x @ W^T  =>  B[k][n] = W[n][k].
// Layout: lane l -> N = l%16, element e -> K = e + 16*(l/16):
// each lane reads 16 contiguous K values of one W row (4x global b128).
// ---------------------------------------------------------------------------
__device__ __forceinline__ v16bf load_w_frag(const float* __restrict__ W,
                                             int ldw, int n0, int k0) {
  const int lane = threadIdx.x & 31;
  const int col  = lane & 15;
  const int half = lane >> 4;
  const float* p = W + (size_t)(n0 + col) * ldw + (k0 + 16 * half);
  const float4 q0 = ((const float4*)p)[0];
  const float4 q1 = ((const float4*)p)[1];
  const float4 q2 = ((const float4*)p)[2];
  const float4 q3 = ((const float4*)p)[3];
  v16bf r;
  r[0]=(__bf16)q0.x; r[1]=(__bf16)q0.y; r[2]=(__bf16)q0.z; r[3]=(__bf16)q0.w;
  r[4]=(__bf16)q1.x; r[5]=(__bf16)q1.y; r[6]=(__bf16)q1.z; r[7]=(__bf16)q1.w;
  r[8]=(__bf16)q2.x; r[9]=(__bf16)q2.y; r[10]=(__bf16)q2.z; r[11]=(__bf16)q2.w;
  r[12]=(__bf16)q3.x; r[13]=(__bf16)q3.y; r[14]=(__bf16)q3.z; r[15]=(__bf16)q3.w;
  return r;
}

// ---------------------------------------------------------------------------
// A-matrix fragment (16x32 MxK, bf16) from a row-major bf16 LDS tile.
// Layout: lane l -> M = l%16; element e -> K = 16*(l/16) + e
// => two ds_load_b128 per lane.
// ---------------------------------------------------------------------------
__device__ __forceinline__ v16bf load_a_frag(const __bf16* __restrict__ base,
                                             int ld, int k0) {
  const int lane = threadIdx.x & 31;
  const int row  = lane & 15;
  const int half = lane >> 4;
  const __bf16* p = base + row * ld + k0 + 16 * half;
  v16bf r;
  ((uint4*)&r)[0] = *(const uint4*)(p);
  ((uint4*)&r)[1] = *(const uint4*)(p + 8);
  return r;
}

__device__ __forceinline__ v8f vbcast(float x) {
  v8f r;
#pragma unroll
  for (int i = 0; i < 8; ++i) r[i] = x;
  return r;
}

// v_exp_f32 + v_rcp_f32 — avoids the IEEE divide expansion
__device__ __forceinline__ float sigmf(float x) {
  return __builtin_amdgcn_rcpf(1.0f + __expf(-x));
}
__device__ __forceinline__ float tanh_fast(float x) {
  return 2.0f * __builtin_amdgcn_rcpf(1.0f + __expf(-2.0f * x)) - 1.0f;
}

// ---------------------------------------------------------------------------
// One block = 16 batch rows, 4 waves. Wave w owns H columns [16w, 16w+16).
// h0/h1 carried in f32 C-layout registers; bf16 copies double-buffered in LDS.
// x tile software-pipelined: next step's global loads issued before compute,
// stored to the alternate xbuf at the top of the next iteration.
// 30 bf16 WMMAs per wave per timestep; 2 barriers per timestep.
// ---------------------------------------------------------------------------
__global__ __launch_bounds__(128, 1) void gru2_fused_kernel(
    const float* __restrict__ state,
    const float* __restrict__ Wih0, const float* __restrict__ Whh0,
    const float* __restrict__ bih0, const float* __restrict__ bhh0,
    const float* __restrict__ Wih1, const float* __restrict__ Whh1,
    const float* __restrict__ bih1, const float* __restrict__ bhh1,
    const float* __restrict__ fc3w, const float* __restrict__ fc3b,
    float* __restrict__ out) {
  constexpr int T = 512, F = 128, H = 64, A = 18;
  constexpr int XLD = 136;  // 128 + 8 bf16 pad (16B-aligned rows, bank stagger)
  constexpr int HLD = 72;   // 64 + 8 bf16 pad
  constexpr int PF  = 6;    // L2-warm prefetch distance (timesteps)

  __shared__ __bf16 xbuf[2][16 * XLD];
  __shared__ __bf16 h0buf[2][16 * HLD];
  __shared__ __bf16 h1buf[2][16 * HLD];
  __shared__ float  hout[16][H + 1];

  const int m0   = blockIdx.x * 16;
  const int wave = threadIdx.x >> 5;
  const int lane = threadIdx.x & 31;
  const int col  = lane & 15;
  const int half = lane >> 4;
  const int c0   = wave * 16;  // this wave's H-column tile

  // zero hidden-state LDS tiles (both parities)
  for (int i = threadIdx.x; i < 16 * HLD; i += 128) {
    h0buf[0][i] = (__bf16)0.0f; h0buf[1][i] = (__bf16)0.0f;
    h1buf[0][i] = (__bf16)0.0f; h1buf[1][i] = (__bf16)0.0f;
  }

  // ---- loop-invariant weight fragments in VGPRs (bf16 B-layout) ----
  v16bf wih0f[3][4], whh0f[3][2], wih1f[3][2], whh1f[3][2];
#pragma unroll
  for (int g = 0; g < 3; ++g) {
    const int n0 = g * 64 + c0;  // gate order r,z,n
#pragma unroll
    for (int kc = 0; kc < 4; ++kc) wih0f[g][kc] = load_w_frag(Wih0, F, n0, kc * 32);
#pragma unroll
    for (int kc = 0; kc < 2; ++kc) {
      whh0f[g][kc] = load_w_frag(Whh0, H, n0, kc * 32);
      wih1f[g][kc] = load_w_frag(Wih1, H, n0, kc * 32);
      whh1f[g][kc] = load_w_frag(Whh1, H, n0, kc * 32);
    }
  }

  // loop-invariant fused biases (N depends only on lane%16)
  const float br0  = bih0[c0 + col]       + bhh0[c0 + col];
  const float bz0  = bih0[64 + c0 + col]  + bhh0[64 + c0 + col];
  const float bxn0 = bih0[128 + c0 + col];
  const float bhn0 = bhh0[128 + c0 + col];
  const float br1  = bih1[c0 + col]       + bhh1[c0 + col];
  const float bz1  = bih1[64 + c0 + col]  + bhh1[64 + c0 + col];
  const float bxn1 = bih1[128 + c0 + col];
  const float bhn1 = bhh1[128 + c0 + col];

  v8f h0r = vbcast(0.0f);  // f32 hidden state, C-layout
  v8f h1r = vbcast(0.0f);

  // per-thread x-tile chunk coordinates (4 x float4 per thread per step)
  int xr[4], xc[4];
#pragma unroll
  for (int ii = 0; ii < 4; ++ii) {
    const int i = threadIdx.x + ii * 128;  // 512 float4 chunks of the 16x128 tile
    xr[ii] = i >> 5;
    xc[ii] = (i & 31) << 2;
  }
  const float* xbase = state + (size_t)m0 * T * F;  // + r*T*F + t*F + c

  // prologue: prefetch tile t=0 into registers
  float4 xf[4];
#pragma unroll
  for (int ii = 0; ii < 4; ++ii)
    xf[ii] = *(const float4*)(xbase + (size_t)xr[ii] * T * F + xc[ii]);

  __syncthreads();

  for (int t = 0; t < T; ++t) {
    const int p = t & 1;

    // ---- publish prefetched x tile (f32 regs -> bf16 LDS, parity p) ----
#pragma unroll
    for (int ii = 0; ii < 4; ++ii) {
      __bf16* d = &xbuf[p][xr[ii] * XLD + xc[ii]];
      d[0] = (__bf16)xf[ii].x; d[1] = (__bf16)xf[ii].y;
      d[2] = (__bf16)xf[ii].z; d[3] = (__bf16)xf[ii].w;
    }
    // ---- issue next step's global loads: latency hidden behind compute ----
    if (t + 1 < T) {
#pragma unroll
      for (int ii = 0; ii < 4; ++ii)
        xf[ii] = *(const float4*)(xbase + (size_t)xr[ii] * T * F +
                                  (size_t)(t + 1) * F + xc[ii]);
    }
    // ---- L2-warm prefetch a few steps ahead (global_prefetch_b8) ----
    if (t + PF < T) {
#pragma unroll
      for (int ii = 0; ii < 4; ++ii)
        __builtin_prefetch(xbase + (size_t)xr[ii] * T * F +
                           (size_t)(t + PF) * F + xc[ii], 0, 1);
    }
    __syncthreads();  // B1: xbuf[p] ready; prev-step h1buf[p] stores visible

    // ---- layer 0: gates = x@Wih0^T + h0@Whh0^T + biases ----
    v16bf ax[4];
#pragma unroll
    for (int kc = 0; kc < 4; ++kc) ax[kc] = load_a_frag(xbuf[p], XLD, kc * 32);
    v16bf ah0[2];
#pragma unroll
    for (int kc = 0; kc < 2; ++kc) ah0[kc] = load_a_frag(h0buf[p], HLD, kc * 32);

    v8f ar  = vbcast(br0);
    v8f az  = vbcast(bz0);
    v8f axn = vbcast(bxn0);
    v8f ahn = vbcast(bhn0);
#pragma unroll
    for (int kc = 0; kc < 4; ++kc) {
      ar  = WMMA_BF16(ax[kc], wih0f[0][kc], ar);
      az  = WMMA_BF16(ax[kc], wih0f[1][kc], az);
      axn = WMMA_BF16(ax[kc], wih0f[2][kc], axn);
    }
#pragma unroll
    for (int kc = 0; kc < 2; ++kc) {
      ar  = WMMA_BF16(ah0[kc], whh0f[0][kc], ar);
      az  = WMMA_BF16(ah0[kc], whh0f[1][kc], az);
      ahn = WMMA_BF16(ah0[kc], whh0f[2][kc], ahn);  // n = tanh(xn + r*hn)
    }

    // ---- GRU update h0 (f32 carry), publish bf16 h0 to parity !p ----
#pragma unroll
    for (int j = 0; j < 8; ++j) {
      const float r = sigmf(ar[j]);
      const float z = sigmf(az[j]);
      const float n = tanh_fast(axn[j] + r * ahn[j]);
      const float h = (1.0f - z) * n + z * h0r[j];
      h0r[j] = h;
      h0buf[1 - p][(j + 8 * half) * HLD + c0 + col] = (__bf16)h;
    }
    __syncthreads();  // B3: new h0 (parity !p) ready for all waves

    // ---- layer 1: gates = h0_new@Wih1^T + h1@Whh1^T + biases ----
    v16bf a1[2], ah1[2];
#pragma unroll
    for (int kc = 0; kc < 2; ++kc) {
      a1[kc]  = load_a_frag(h0buf[1 - p], HLD, kc * 32);
      ah1[kc] = load_a_frag(h1buf[p], HLD, kc * 32);
    }
    v8f cr  = vbcast(br1);
    v8f cz  = vbcast(bz1);
    v8f cxn = vbcast(bxn1);
    v8f chn = vbcast(bhn1);
#pragma unroll
    for (int kc = 0; kc < 2; ++kc) {
      cr  = WMMA_BF16(a1[kc], wih1f[0][kc], cr);
      cz  = WMMA_BF16(a1[kc], wih1f[1][kc], cz);
      cxn = WMMA_BF16(a1[kc], wih1f[2][kc], cxn);
      cr  = WMMA_BF16(ah1[kc], whh1f[0][kc], cr);
      cz  = WMMA_BF16(ah1[kc], whh1f[1][kc], cz);
      chn = WMMA_BF16(ah1[kc], whh1f[2][kc], chn);
    }

#pragma unroll
    for (int j = 0; j < 8; ++j) {
      const float r = sigmf(cr[j]);
      const float z = sigmf(cz[j]);
      const float n = tanh_fast(cxn[j] + r * chn[j]);
      const float h = (1.0f - z) * n + z * h1r[j];
      h1r[j] = h;
      h1buf[1 - p][(j + 8 * half) * HLD + c0 + col] = (__bf16)h;
    }
    // next iteration's B1 orders these stores before the next h1buf reads
  }

  // ---- tail: actions = relu(h1_last) @ fc3w^T + fc3b ----
#pragma unroll
  for (int j = 0; j < 8; ++j)
    hout[j + 8 * half][c0 + col] = fmaxf(h1r[j], 0.0f);
  __syncthreads();

  for (int i = threadIdx.x; i < 16 * A; i += 128) {
    const int r = i / A;
    const int a = i % A;
    float acc = fc3b[a];
#pragma unroll
    for (int k = 0; k < H; ++k) acc += hout[r][k] * fc3w[a * H + k];
    out[(size_t)(m0 + r) * A + a] = acc;
  }
}

extern "C" void kernel_launch(void* const* d_in, const int* in_sizes, int n_in,
                              void* d_out, int out_size, void* d_ws, size_t ws_size,
                              hipStream_t stream) {
  (void)in_sizes; (void)n_in; (void)d_ws; (void)ws_size; (void)out_size;
  const float* state = (const float*)d_in[0];
  const float* Wih0  = (const float*)d_in[1];
  const float* Whh0  = (const float*)d_in[2];
  const float* bih0  = (const float*)d_in[3];
  const float* bhh0  = (const float*)d_in[4];
  const float* Wih1  = (const float*)d_in[5];
  const float* Whh1  = (const float*)d_in[6];
  const float* bih1  = (const float*)d_in[7];
  const float* bhh1  = (const float*)d_in[8];
  const float* fc3w  = (const float*)d_in[9];
  const float* fc3b  = (const float*)d_in[10];

  // B=1024 -> 64 blocks of 16 batch rows; 128 threads = 4 wave32 per block.
  gru2_fused_kernel<<<64, 128, 0, stream>>>(state, Wih0, Whh0, bih0, bhh0,
                                            Wih1, Whh1, bih1, bhh1,
                                            fc3w, fc3b, (float*)d_out);
}
